// LlamaSmartKVAttention_54631984005702
// MI455X (gfx1250) — matrix operand order repaired and checked by reading, not compiled, not run on hardware
//
#include <hip/hip_runtime.h>
#include <hip/hip_bf16.h>
#include <math.h>

// ---------------------------------------------------------------------------
// MI455X / gfx1250 implementation of LlamaSmartKVAttention.
// Memory-bound problem (~435MB/call vs 23.3TB/s); all GEMM-like work uses
// v_wmma_f32_16x16x32_f16 with f32 accumulation, converting f32/int32 inputs
// to f16 in-register (no extra global traffic).
// ---------------------------------------------------------------------------

typedef __attribute__((ext_vector_type(16))) _Float16 v16h;
typedef __attribute__((ext_vector_type(8)))  _Float16 h8;
typedef __attribute__((ext_vector_type(4)))  _Float16 h4;
typedef __attribute__((ext_vector_type(8)))  float    v8f;

#define B_    4
#define S_    16
#define H_    4096
#define NH_   32
#define KVH_  8
#define DH_   128
#define PAST_ 8192
#define NCH_  17            // 16 past chunks of 512 keys + 1 current chunk (16 keys)
#define NEG_INF_F (-3.402823466e38f)

// ---- WMMA fragment loaders (ISA 7.12.2 layouts, wave32) --------------------
// A 16x32 f16: lane L holds row (L&15); halfs[0:8) at col 8*(L>>4), halfs[8:16) at +16.
__device__ __forceinline__ v16h frag_a_ld(const _Float16* base, int ld, int lane) {
  const _Float16* p = base + (lane & 15) * ld + ((lane >> 4) << 3);
  h8 lo = *(const h8*)(p);
  h8 hi = *(const h8*)(p + 16);
  return __builtin_shufflevector(lo, hi, 0,1,2,3,4,5,6,7,8,9,10,11,12,13,14,15);
}
// B 32x16 f16 from N-major memory (each N row has contiguous K):
// lane L holds col N=(L&15); halfs[0:8) at K 8*(L>>4), halfs[8:16) at +16.
__device__ __forceinline__ v16h frag_b_ld(const _Float16* base, int ld, int lane) {
  const _Float16* p = base + (lane & 15) * ld + ((lane >> 4) << 3);
  h8 lo = *(const h8*)(p);
  h8 hi = *(const h8*)(p + 16);
  return __builtin_shufflevector(lo, hi, 0,1,2,3,4,5,6,7,8,9,10,11,12,13,14,15);
}

__device__ __forceinline__ float red_max16(float v) {
#pragma unroll
  for (int m = 1; m <= 8; m <<= 1) v = fmaxf(v, __shfl_xor(v, m, 32));
  return v;
}
__device__ __forceinline__ float red_sum16(float v) {
#pragma unroll
  for (int m = 1; m <= 8; m <<= 1) v += __shfl_xor(v, m, 32);
  return v;
}

// ---------------------------------------------------------------------------
// Skinny GEMM: out(64 x N) = A(64 x 4096) * W(N x 4096)^T, f16 WMMA, f32 acc.
// mode 0: plain row-major out[m*4096 + n]                (final Wo projection)
// mode 1: head layout ((b*nh+h)*16+s)*128+d, fused RoPE  (Q, K)
// mode 2: head layout, no RoPE                           (V)
// grid: (N/128, 4). block: 256 (8 waves; wave w owns N cols [w*16, w*16+16)).
// ---------------------------------------------------------------------------
__global__ __launch_bounds__(256)
void gemm_f16w(const float* __restrict__ A, const float* __restrict__ W,
               const float* __restrict__ cosT, const float* __restrict__ sinT,
               float* __restrict__ out, int mode, int nh) {
  __shared__ __align__(16) _Float16 As[16 * 64];
  __shared__ __align__(16) _Float16 Bs[128 * 64];
  __shared__ float Cs[16 * 132];
  const int t    = threadIdx.x;
  const int lane = t & 31;
  const int w    = t >> 5;
  const int m0   = blockIdx.y * 16;
  const int n0   = blockIdx.x * 128;

  v8f c = {0.f, 0.f, 0.f, 0.f, 0.f, 0.f, 0.f, 0.f};

  for (int kk = 0; kk < H_; kk += 64) {
    { // stage A tile 16x64 (f32 -> f16)
      int i = t * 4;                 // 0..1020
      int r = i >> 6, ci = i & 63;
      float4 av = *(const float4*)(A + (long)(m0 + r) * H_ + kk + ci);
      h4 hv = { (_Float16)av.x, (_Float16)av.y, (_Float16)av.z, (_Float16)av.w };
      *(h4*)(As + r * 64 + ci) = hv;
    }
    { // stage W tile 128x64 (f32 -> f16), W is (N,K) row-major -> N-major f16
      int row = t >> 1;
      int cb  = (t & 1) * 32;
      const float* wp = W + (long)(n0 + row) * H_ + kk + cb;
#pragma unroll
      for (int jj = 0; jj < 8; ++jj) {
        float4 bv = *(const float4*)(wp + jj * 4);
        h4 hv = { (_Float16)bv.x, (_Float16)bv.y, (_Float16)bv.z, (_Float16)bv.w };
        *(h4*)(Bs + row * 64 + cb + jj * 4) = hv;
      }
    }
    __syncthreads();
#pragma unroll
    for (int kc = 0; kc < 64; kc += 32) {
      v16h a = frag_a_ld(As + kc, 64, lane);
      v16h b = frag_b_ld(Bs + w * 16 * 64 + kc, 64, lane);
      c = __builtin_amdgcn_wmma_f32_16x16x32_f16(false, a, false, b,
                                                 (short)0, c, false, false);
    }
    __syncthreads();
  }

  // stage C tile to LDS (C layout: m = r + 8*(lane>>4), n = lane&15)
#pragma unroll
  for (int r = 0; r < 8; ++r)
    Cs[(r + ((lane >> 4) << 3)) * 132 + w * 16 + (lane & 15)] = c[r];
  __syncthreads();

  // epilogue: each thread handles (m = t>>4, 8 consecutive n)
  const int m  = t >> 4;
  const int db = (t & 15) * 8;
  const int gm = m0 + m;
  const int bb = gm >> 4, s = gm & 15;
#pragma unroll
  for (int j = 0; j < 8; ++j) {
    int nl = db + j;
    float x = Cs[m * 132 + nl];
    if (mode == 0) {
      out[(long)gm * H_ + n0 + nl] = x;
    } else {
      int ng = n0 + nl;
      int h  = ng >> 7, d = ng & 127;   // N-tile (128) is head aligned -> nl == d
      float val = x;
      if (mode == 1) {
        float partner = Cs[m * 132 + (nl ^ 64)];
        float rot = (d < 64) ? -partner : partner;
        val = x * cosT[s * DH_ + d] + rot * sinT[s * DH_ + d];
      }
      out[(((long)bb * nh + h) * S_ + s) * DH_ + d] = val;
    }
  }
}

// ---------------------------------------------------------------------------
// Flash-attention partials with split-K over keys + fused int dequant.
// grid: B*KVH*NCH_ workgroups of 128 threads (4 waves = 4 GQA heads).
// chunk < 16: 512 past keys (dequantized from codes), chunk 16: the 16
// current (roped) keys with causal mask.
// ---------------------------------------------------------------------------
__global__ __launch_bounds__(128)
void attn_partial(const float* __restrict__ qg, const float* __restrict__ kcur,
                  const float* __restrict__ vcur,
                  const int* __restrict__ kcodes, const int* __restrict__ vcodes,
                  const float* __restrict__ kscales, const float* __restrict__ vscales,
                  const int* __restrict__ bitsidx,
                  float* __restrict__ Opart, float* __restrict__ mpart,
                  float* __restrict__ lpart) {
  __shared__ __align__(16) _Float16 Qs[64 * 128];       // 16KB
  __shared__ __align__(16) _Float16 Tile[4][32 * 128];  // 8KB per wave (K, then V^T)
  __shared__ __align__(16) _Float16 Pt[4][16 * 32];     // 1KB per wave (softmax probs)

  const int t = threadIdx.x, lane = t & 31, wv = t >> 5;
  int blk = blockIdx.x;
  const int ch  = blk % NCH_;  blk /= NCH_;
  const int kvh = blk & 7;
  const int b   = blk >> 3;

  // stage Q (4 heads x 16 x 128) as f16
  const float* qbase = qg + (long)(b * NH_ + kvh * 4) * (S_ * DH_);
  for (int i = t; i < 64 * 128; i += 128) Qs[i] = (_Float16)qbase[i];
  __syncthreads();

  const int hl = wv;                       // head_local 0..3
  v16h qa[4];
#pragma unroll
  for (int kc = 0; kc < 4; ++kc)
    qa[kc] = frag_a_ld(Qs + hl * 16 * 128 + kc * 32, 128, lane);

  float mr[8], lr[8];
  v8f O[8];
#pragma unroll
  for (int r = 0; r < 8; ++r) { mr[r] = NEG_INF_F; lr[r] = 0.f; }
#pragma unroll
  for (int dt = 0; dt < 8; ++dt) O[dt] = (v8f){0.f,0.f,0.f,0.f,0.f,0.f,0.f,0.f};

  _Float16* KT = Tile[wv];
  _Float16* PT = Pt[wv];
  const long kvbase  = (long)(b * KVH_ + kvh) * PAST_;
  const float qscale = 0.08838834764831845f;   // 1/sqrt(128)
  const bool past = (ch < 16);
  const int key0 = past ? ch * 512 : 0;
  const int nblocks = past ? 16 : 1;

  for (int bk = 0; bk < nblocks; ++bk) {
    // ---------------- dequant K tile (32 keys x 128 d, key-major) ----------
    if (past) {
      int keyb = key0 + bk * 32;
      const int4* cp = (const int4*)(kcodes + (kvbase + keyb) * DH_);
#pragma unroll 4
      for (int j = 0; j < 32; ++j) {
        int gi = j * 32 + lane;
        int4 cc = cp[gi];
        int flat = gi << 2;
        int tkey = keyb + (flat >> 7);
        int bits = (0x08040302 >> (bitsidx[tkey] * 8)) & 0xff;
        int msk  = (1 << bits) - 1;
        float hv = (float)(1 << (bits - 1));
        float sc = kscales[kvbase + tkey];
        h4 o = { (_Float16)(((float)(cc.x & msk) - hv) * sc),
                 (_Float16)(((float)(cc.y & msk) - hv) * sc),
                 (_Float16)(((float)(cc.z & msk) - hv) * sc),
                 (_Float16)(((float)(cc.w & msk) - hv) * sc) };
        *(h4*)(KT + flat) = o;
      }
    } else {
      const float* kc_ = kcur + (long)(b * KVH_ + kvh) * (S_ * DH_);
#pragma unroll 4
      for (int j = 0; j < 64; ++j) {
        int idx = j * 32 + lane;            // 16 keys x 128
        KT[idx] = (_Float16)kc_[idx];
      }
    }
    asm volatile("s_wait_dscnt 0x0" ::: "memory");

    // ---------------- scores: q(16x128) . k^T ------------------------------
    v8f sc0 = (v8f){0.f,0.f,0.f,0.f,0.f,0.f,0.f,0.f};
    v8f sc1;
#pragma unroll
    for (int kc = 0; kc < 4; ++kc) {
      v16h bkf = frag_b_ld(KT + kc * 32, 128, lane);
      sc0 = __builtin_amdgcn_wmma_f32_16x16x32_f16(false, qa[kc], false, bkf,
                                                   (short)0, sc0, false, false);
    }
    if (past) {
      sc1 = (v8f){0.f,0.f,0.f,0.f,0.f,0.f,0.f,0.f};
#pragma unroll
      for (int kc = 0; kc < 4; ++kc) {
        v16h bkf = frag_b_ld(KT + 16 * 128 + kc * 32, 128, lane);
        sc1 = __builtin_amdgcn_wmma_f32_16x16x32_f16(false, qa[kc], false, bkf,
                                                     (short)0, sc1, false, false);
      }
#pragma unroll
      for (int r = 0; r < 8; ++r) { sc0[r] *= qscale; sc1[r] *= qscale; }
    } else {
#pragma unroll
      for (int r = 0; r < 8; ++r) {
        int row = r + ((lane >> 4) << 3);
        sc0[r] = ((lane & 15) > row) ? NEG_INF_F : sc0[r] * qscale;
        sc1[r] = NEG_INF_F;
      }
    }

    // ---------------- online softmax (rows live in 16-lane groups) ---------
    float pf0[8], pf1[8], alpha[8];
#pragma unroll
    for (int r = 0; r < 8; ++r) {
      float rm   = red_max16(fmaxf(sc0[r], sc1[r]));
      float mnew = fmaxf(mr[r], rm);
      alpha[r]   = __expf(mr[r] - mnew);
      pf0[r]     = __expf(sc0[r] - mnew);
      pf1[r]     = __expf(sc1[r] - mnew);
      lr[r]      = lr[r] * alpha[r] + red_sum16(pf0[r] + pf1[r]);
      mr[r]      = mnew;
    }
#pragma unroll
    for (int dt = 0; dt < 8; ++dt)
#pragma unroll
      for (int r = 0; r < 8; ++r) O[dt][r] *= alpha[r];

    // stage probs as f16 A-source [s][key], ld=32
#pragma unroll
    for (int r = 0; r < 8; ++r) {
      int row = r + ((lane >> 4) << 3);
      PT[row * 32 + (lane & 15)]      = (_Float16)pf0[r];
      PT[row * 32 + 16 + (lane & 15)] = (_Float16)pf1[r];
    }

    // ---------------- dequant V, transposed into KT as [d][key], ld=32 -----
    if (past) {
      int keyb = key0 + bk * 32;
      const int4* cp = (const int4*)(vcodes + (kvbase + keyb) * DH_);
#pragma unroll 4
      for (int j = 0; j < 32; ++j) {
        int gi = j * 32 + lane;
        int4 cc = cp[gi];
        int flat = gi << 2;
        int kl   = flat >> 7;
        int d0   = flat & 127;
        int tkey = keyb + kl;
        int bits = (0x08040302 >> (bitsidx[tkey] * 8)) & 0xff;
        int msk  = (1 << bits) - 1;
        float hv = (float)(1 << (bits - 1));
        float sc = vscales[kvbase + tkey];
        KT[(d0 + 0) * 32 + kl] = (_Float16)(((float)(cc.x & msk) - hv) * sc);
        KT[(d0 + 1) * 32 + kl] = (_Float16)(((float)(cc.y & msk) - hv) * sc);
        KT[(d0 + 2) * 32 + kl] = (_Float16)(((float)(cc.z & msk) - hv) * sc);
        KT[(d0 + 3) * 32 + kl] = (_Float16)(((float)(cc.w & msk) - hv) * sc);
      }
    } else {
      const float* vc_ = vcur + (long)(b * KVH_ + kvh) * (S_ * DH_);
#pragma unroll 4
      for (int j = 0; j < 128; ++j) {
        int i = j * 32 + lane;              // [d][key] over 128x32
        int key = i & 31, d = i >> 5;
        KT[i] = (key < 16) ? (_Float16)vc_[key * DH_ + d] : (_Float16)0.f;
      }
    }
    asm volatile("s_wait_dscnt 0x0" ::: "memory");

    // ---------------- O += P(16x32) x V(32x128) ----------------------------
    v16h pa = frag_a_ld(PT, 32, lane);
#pragma unroll
    for (int dt = 0; dt < 8; ++dt) {
      v16h bv = frag_b_ld(KT + dt * 16 * 32, 32, lane);
      O[dt] = __builtin_amdgcn_wmma_f32_16x16x32_f16(false, pa, false, bv,
                                                     (short)0, O[dt], false, false);
    }
  }

  // ---------------- write partial (O, m, l) --------------------------------
  const int h = kvh * 4 + hl;
  const long obase = ((long)((b * NH_ + h) * NCH_ + ch)) * (16 * 128);
#pragma unroll
  for (int dt = 0; dt < 8; ++dt)
#pragma unroll
    for (int r = 0; r < 8; ++r)
      Opart[obase + (long)(r + ((lane >> 4) << 3)) * 128 + dt * 16 + (lane & 15)] = O[dt][r];
  if ((lane & 15) == 0) {
    int mb = ((b * NH_ + h) * NCH_ + ch) * 16;
#pragma unroll
    for (int r = 0; r < 8; ++r) {
      int row = r + ((lane >> 4) << 3);
      mpart[mb + row] = mr[r];
      lpart[mb + row] = lr[r];
    }
  }
}

// ---------------------------------------------------------------------------
// Combine split-K partials -> attn_out in (b,s,h,d) row-major (64 x 4096).
// grid: B*NH blocks of 128 threads (thread = (s, 16-wide d group)).
// ---------------------------------------------------------------------------
__global__ __launch_bounds__(128)
void attn_combine(const float* __restrict__ Opart, const float* __restrict__ mpart,
                  const float* __restrict__ lpart, float* __restrict__ attn_out) {
  const int bh = blockIdx.x;
  const int b = bh >> 5, h = bh & 31;
  const int t = threadIdx.x;
  const int s = t >> 3, dg = t & 7;
  const int mb = bh * NCH_ * 16 + s;

  float M = NEG_INF_F;
#pragma unroll
  for (int p = 0; p < NCH_; ++p) M = fmaxf(M, mpart[mb + p * 16]);

  float L = 0.f;
  float acc[16];
#pragma unroll
  for (int i = 0; i < 16; ++i) acc[i] = 0.f;

  const long ob = (long)bh * NCH_ * 16 * 128;
  for (int p = 0; p < NCH_; ++p) {
    float wgt = __expf(mpart[mb + p * 16] - M);
    L += wgt * lpart[mb + p * 16];
    if (wgt > 0.f) {
      const float* op = Opart + ob + (long)(p * 16 + s) * 128 + dg * 16;
#pragma unroll
      for (int i = 0; i < 16; ++i) acc[i] += wgt * op[i];
    }
  }
  float inv = 1.0f / L;
  float* outp = attn_out + ((long)(b * S_ + s) * NH_ + h) * DH_ + dg * 16;
#pragma unroll
  for (int i = 0; i < 16; ++i) outp[i] = acc[i] * inv;
}

// ---------------------------------------------------------------------------
extern "C" void kernel_launch(void* const* d_in, const int* in_sizes, int n_in,
                              void* d_out, int out_size, void* d_ws, size_t ws_size,
                              hipStream_t stream) {
  (void)in_sizes; (void)n_in; (void)out_size; (void)ws_size;
  const float* hs      = (const float*)d_in[0];
  const float* Wq      = (const float*)d_in[1];
  const float* Wk      = (const float*)d_in[2];
  const float* Wv      = (const float*)d_in[3];
  const float* Wo      = (const float*)d_in[4];
  const float* cosT    = (const float*)d_in[5];
  const float* sinT    = (const float*)d_in[6];
  const int*   kcodes  = (const int*)d_in[7];
  const int*   vcodes  = (const int*)d_in[8];
  const float* kscales = (const float*)d_in[9];
  const float* vscales = (const float*)d_in[10];
  const int*   bitsidx = (const int*)d_in[11];

  float* ws      = (float*)d_ws;
  float* q_ws    = ws;                                       // 64*4096
  float* k_ws    = q_ws + (long)64 * 4096;                   // 4*8*16*128
  float* v_ws    = k_ws + (long)B_ * KVH_ * S_ * DH_;
  float* attn_ws = v_ws + (long)B_ * KVH_ * S_ * DH_;        // 64*4096
  float* Opart   = attn_ws + (long)64 * 4096;                // B*NH*17*16*128
  float* mpart   = Opart + (long)B_ * NH_ * NCH_ * 16 * 128;
  float* lpart   = mpart + (long)B_ * NH_ * NCH_ * 16;

  dim3 blk(256);
  gemm_f16w<<<dim3(32, 4), blk, 0, stream>>>(hs, Wq, cosT, sinT, q_ws, 1, NH_);
  gemm_f16w<<<dim3(8, 4),  blk, 0, stream>>>(hs, Wk, cosT, sinT, k_ws, 1, KVH_);
  gemm_f16w<<<dim3(8, 4),  blk, 0, stream>>>(hs, Wv, cosT, sinT, v_ws, 2, KVH_);
  attn_partial<<<dim3(B_ * KVH_ * NCH_), dim3(128), 0, stream>>>(
      q_ws, k_ws, v_ws, kcodes, vcodes, kscales, vscales, bitsidx,
      Opart, mpart, lpart);
  attn_combine<<<dim3(B_ * NH_), dim3(128), 0, stream>>>(Opart, mpart, lpart, attn_ws);
  gemm_f16w<<<dim3(32, 4), blk, 0, stream>>>(attn_ws, Wo, nullptr, nullptr,
                                             (float*)d_out, 0, 0);
}
